// AttentiveItemToVec_4896262717596
// MI455X (gfx1250) — matrix operand order repaired and checked by reading, not compiled, not run on hardware
//
#include <hip/hip_runtime.h>
#include <hip/hip_bf16.h>

typedef __bf16 bf16;
typedef __attribute__((ext_vector_type(16))) __bf16 v16bf;
typedef __attribute__((ext_vector_type(8)))  float  v8f;
typedef __attribute__((ext_vector_type(4)))  int    v4i;

#define B_   128
#define NT_  512
#define NC_  512
#define EMB_ 256
#define DK_  60
#define DP_  64
#define ROWS_ 32
#define EPS_ 1e-6f

// ---- CDNA5 async global->LDS (ASYNCcnt path), guarded ----------------------
#if defined(__gfx1250__) && __has_builtin(__builtin_amdgcn_global_load_async_to_lds_b128)
#define HAS_ASYNC_LDS 1
#else
#define HAS_ASYNC_LDS 0
#endif

#if HAS_ASYNC_LDS
typedef __attribute__((address_space(1))) v4i* gv4i_p;  // global int4*
typedef __attribute__((address_space(3))) v4i* lv4i_p;  // LDS int4*
#endif

__device__ __forceinline__ void async_cp16(void* lds_dst, const void* gsrc) {
#if HAS_ASYNC_LDS
  __builtin_amdgcn_global_load_async_to_lds_b128(
      (gv4i_p)(unsigned long long)gsrc,
      (lv4i_p)(unsigned int)(unsigned long long)lds_dst,
      0, 0);
#else
  *(uint4*)lds_dst = *(const uint4*)gsrc;
#endif
}
__device__ __forceinline__ void async_wait_all() {
#if HAS_ASYNC_LDS
#if __has_builtin(__builtin_amdgcn_s_wait_asynccnt)
  __builtin_amdgcn_s_wait_asynccnt(0);
#else
  asm volatile("s_wait_asynccnt 0x0" ::: "memory");
#endif
#endif
}

// ---- CDNA5 WMMA fragment packing (ISA 7.12.2, wave32) -----------------------
// A (16x32 bf16, row-major src, stride ldm): lane&15 = M; lane>>4 selects K-half.
__device__ __forceinline__ v16bf frag_a(const bf16* base, int ldm, int lane) {
  const int m = lane & 15, kh = (lane >> 4) << 3;
  const bf16* p = base + m * ldm + kh;
  union { v16bf v; bf16 e[16]; } u;
#pragma unroll
  for (int e = 0; e < 8; ++e) { u.e[e] = p[e]; u.e[e + 8] = p[16 + e]; }
  return u.v;
}
// B (32x16 bf16) read from an N x K row-major matrix W (B[k][n] = W[n][k]):
__device__ __forceinline__ v16bf frag_b_nk(const bf16* w, int ldw, int lane) {
  const int n = lane & 15, kh = (lane >> 4) << 3;
  const bf16* p = w + n * ldw + kh;
  union { v16bf v; bf16 e[16]; } u;
#pragma unroll
  for (int e = 0; e < 8; ++e) { u.e[e] = p[e]; u.e[e + 8] = p[16 + e]; }
  return u.v;
}
__device__ __forceinline__ v8f wmma_bf16(v16bf a, v16bf b, v8f c) {
  return __builtin_amdgcn_wmma_f32_16x16x32_bf16(false, a, false, b, (short)0, c,
                                                 false, false);
}
// C/D layout: lane&15 = N ; row = reg + 8*(lane>>4)

// ---- kernel 0: weights -> bf16, pad 60 -> 64 with zeros ---------------------
__global__ __launch_bounds__(256) void k_prep(
    const float* At_w, const float* Ac_w, const float* Bc_w, const float* R_w,
    bf16* WA, bf16* WK, bf16* WV, bf16* WR) {
  const int gid = blockIdx.x * 256 + threadIdx.x;
  if (gid < 3 * DP_ * EMB_) {
    const int which = gid / (DP_ * EMB_);
    const int pos = gid % (DP_ * EMB_);
    const int r = pos >> 8, c = pos & 255;
    const float* src = (which == 0) ? At_w : ((which == 1) ? Ac_w : Bc_w);
    bf16* dst = (which == 0) ? WA : ((which == 1) ? WK : WV);
    dst[pos] = (r < DK_) ? (bf16)src[r * EMB_ + c] : (bf16)0.0f;
  } else {
    const int pos = gid - 3 * DP_ * EMB_;
    if (pos < EMB_ * DP_) {
      const int e = pos >> 6, d = pos & 63;
      WR[pos] = (d < DK_) ? (bf16)R_w[e * DK_ + d] : (bf16)0.0f;
    }
  }
}

// ---- kernel 1: gather + Q projection + |q| ----------------------------------
__global__ __launch_bounds__(256) void k_proj_q(
    const int* __restrict__ titems, const float* __restrict__ tvec,
    const bf16* __restrict__ WAg, const float* __restrict__ At_b,
    bf16* __restrict__ Qb, float* __restrict__ qn) {
  extern __shared__ char smem[];
  bf16*  X   = (bf16*)smem;               // 32 x 256 bf16 = 16384 B
  bf16*  W   = (bf16*)(smem + 16384);     // 64 x 256 bf16 = 32768 B
  float* acc = (float*)(smem + 49152);    // 32 f32 = 128 B
  float* bsl = (float*)(smem + 49280);    // 64 f32 = 256 B
  const int tid = threadIdx.x;
  const int b  = blockIdx.x >> 4;
  const int t0 = (blockIdx.x & 15) * ROWS_;
  if (tid < 32) acc[tid] = 0.0f;
  if (tid >= 192 && tid < 256) bsl[tid - 192] = (tid - 192 < DK_) ? At_b[tid - 192] : 0.0f;
  // weights: async global -> LDS (no VGPR round-trip)
  for (int i = tid; i < 2048; i += 256)
    async_cp16((char*)W + i * 16, (const char*)WAg + i * 16);
  // gather rows -> bf16 LDS (conversion overlaps the async weight fill)
  for (int i = tid; i < ROWS_ * (EMB_ / 4); i += 256) {
    const int r = i >> 6, c4 = i & 63;
    const int idx = titems[b * NT_ + t0 + r];
    const float4 f = ((const float4*)(tvec + (size_t)idx * EMB_))[c4];
    bf16* xp = X + r * EMB_ + c4 * 4;
    xp[0] = (bf16)f.x; xp[1] = (bf16)f.y; xp[2] = (bf16)f.z; xp[3] = (bf16)f.w;
  }
  async_wait_all();
  __syncthreads();
  const int wave = tid >> 5, lane = tid & 31;
  const int m0 = (wave >> 2) * 16, n0 = (wave & 3) * 16;
  v8f c = {};
#pragma unroll
  for (int kk = 0; kk < EMB_; kk += 32) {
    v16bf a  = frag_a(X + m0 * EMB_ + kk, EMB_, lane);
    v16bf bb = frag_b_nk(W + n0 * EMB_ + kk, EMB_, lane);
    c = wmma_bf16(a, bb, c);
  }
  const int n = n0 + (lane & 15), kh = (lane >> 4) * 8;
  const float bn = bsl[n];
  bf16* qrow = Qb + ((size_t)b * NT_ + t0) * DP_;
#pragma unroll
  for (int r = 0; r < 8; ++r) {
    const int m = m0 + kh + r;
    const float v = c[r] + bn;
    qrow[m * DP_ + n] = (bf16)v;
    atomicAdd(&acc[m], v * v);            // ds_add_f32
  }
  __syncthreads();
  if (tid < 32) qn[(size_t)b * NT_ + t0 + tid] = sqrtf(acc[tid]);
}

// ---- kernel 2: gather + K,V projections (shared gather) + |k| ---------------
__global__ __launch_bounds__(256) void k_proj_kv(
    const int* __restrict__ citems, const float* __restrict__ cvec,
    const bf16* __restrict__ WKg, const bf16* __restrict__ WVg,
    const float* __restrict__ Ac_b, const float* __restrict__ Bc_b,
    bf16* __restrict__ Kb, bf16* __restrict__ Vt, float* __restrict__ kn) {
  extern __shared__ char smem[];
  bf16*  X   = (bf16*)smem;               // 16384 B
  bf16*  WK  = (bf16*)(smem + 16384);     // 32768 B
  bf16*  WV  = (bf16*)(smem + 49152);     // 32768 B
  float* acc = (float*)(smem + 81920);    // 128 B
  float* bkl = (float*)(smem + 82048);    // 256 B
  float* bvl = (float*)(smem + 82304);    // 256 B
  const int tid = threadIdx.x;
  const int b   = blockIdx.x >> 4;
  const int c0g = (blockIdx.x & 15) * ROWS_;
  if (tid < 32) acc[tid] = 0.0f;
  if (tid >= 192 && tid < 256) {
    const int j = tid - 192;
    bkl[j] = (j < DK_) ? Ac_b[j] : 0.0f;
    bvl[j] = (j < DK_) ? Bc_b[j] : 0.0f;
  }
  for (int i = tid; i < 2048; i += 256)
    async_cp16((char*)WK + i * 16, (const char*)WKg + i * 16);
  for (int i = tid; i < 2048; i += 256)
    async_cp16((char*)WV + i * 16, (const char*)WVg + i * 16);
  for (int i = tid; i < ROWS_ * (EMB_ / 4); i += 256) {
    const int r = i >> 6, c4 = i & 63;
    const int idx = citems[b * NC_ + c0g + r];
    const float4 f = ((const float4*)(cvec + (size_t)idx * EMB_))[c4];
    bf16* xp = X + r * EMB_ + c4 * 4;
    xp[0] = (bf16)f.x; xp[1] = (bf16)f.y; xp[2] = (bf16)f.z; xp[3] = (bf16)f.w;
  }
  async_wait_all();
  __syncthreads();
  const int wave = tid >> 5, lane = tid & 31;
  const int m0 = (wave >> 2) * 16, n0 = (wave & 3) * 16;
  v8f ck = {}, cv = {};
#pragma unroll
  for (int kk = 0; kk < EMB_; kk += 32) {
    v16bf a  = frag_a(X + m0 * EMB_ + kk, EMB_, lane);
    v16bf bk = frag_b_nk(WK + n0 * EMB_ + kk, EMB_, lane);
    v16bf bv = frag_b_nk(WV + n0 * EMB_ + kk, EMB_, lane);
    ck = wmma_bf16(a, bk, ck);
    cv = wmma_bf16(a, bv, cv);
  }
  const int n = n0 + (lane & 15), kh = (lane >> 4) * 8;
  const float bkn = bkl[n];
  const float bvn = bvl[n];
  bf16* krow = Kb + ((size_t)b * NC_ + c0g) * DP_;  // row-major (c, d)
  bf16* vtb  = Vt + (size_t)b * DP_ * NC_;          // transposed (d, c)
  union { uint4 u; bf16 h[8]; } vt;
#pragma unroll
  for (int r = 0; r < 8; ++r) {
    const int m = m0 + kh + r;
    const float kv = ck[r] + bkn;
    krow[m * DP_ + n] = (bf16)kv;
    atomicAdd(&acc[m], kv * kv);
    vt.h[r] = (bf16)(cv[r] + bvn);
  }
  // 8 consecutive context positions -> one 16B store into transposed V
  *(uint4*)(vtb + (size_t)n * NC_ + c0g + m0 + kh) = vt.u;
  __syncthreads();
  if (tid < 32) kn[(size_t)b * NC_ + c0g + tid] = sqrtf(acc[tid]);
}

// ---- kernel 3: fused attention + output projection --------------------------
__global__ __launch_bounds__(256) void k_att(
    const bf16* __restrict__ Qb, const bf16* __restrict__ Kb,
    const bf16* __restrict__ Vt, const float* __restrict__ qn,
    const float* __restrict__ kn, const float* __restrict__ pos_bias,
    const bf16* __restrict__ WRg, const float* __restrict__ R_b,
    float* __restrict__ out) {
  extern __shared__ char smem[];
  bf16*  Kl  = (bf16*)(smem);             // 512 x 64  = 65536 B
  bf16*  Vtl = (bf16*)(smem + 65536);     // 64 x 512  = 65536 B
  float* Sl  = (float*)(smem + 131072);   // 32 x 512  = 65536 B
  bf16*  Pl  = (bf16*)(smem + 196608);    // 32 x 512  = 32768 B
  bf16*  Ql  = (bf16*)(smem + 229376);    // 32 x 64   = 4096 B
  bf16*  WRl = (bf16*)(smem + 233472);    // 256 x 64  = 32768 B
  bf16*  Ol  = (bf16*)(smem + 266240);    // 32 x 64   = 4096 B
  float* knl = (float*)(smem + 270336);   // 2048 B
  float* pbl = (float*)(smem + 272384);   // 2048 B
  float* qnl = (float*)(smem + 274432);   // 128 B
  float* rbl = (float*)(smem + 274560);   // 1024 B   (total 275584 B)

  const int tid = threadIdx.x;
  const int b  = blockIdx.x >> 4;
  const int t0 = (blockIdx.x & 15) * ROWS_;
  const int wave = tid >> 5, lane = tid & 31;

  // bulk staging: async global -> LDS (ASYNCcnt), no VGPR round trips
  { const char* s = (const char*)(Kb + (size_t)b * NC_ * DP_);
    for (int i = tid; i < 4096; i += 256) async_cp16((char*)Kl + i * 16, s + i * 16); }
  { const char* s = (const char*)(Vt + (size_t)b * DP_ * NC_);
    for (int i = tid; i < 4096; i += 256) async_cp16((char*)Vtl + i * 16, s + i * 16); }
  { const char* s = (const char*)(Qb + ((size_t)b * NT_ + t0) * DP_);
    async_cp16((char*)Ql + tid * 16, s + tid * 16); }
  { const char* s = (const char*)WRg;
    for (int i = tid; i < 2048; i += 256) async_cp16((char*)WRl + i * 16, s + i * 16); }
  for (int i = tid; i < NC_; i += 256) {
    knl[i] = kn[(size_t)b * NC_ + i];
    pbl[i] = pos_bias[i];
  }
  for (int i = tid; i < EMB_; i += 256) rbl[i] = R_b[i];
  if (tid < 32) qnl[tid] = qn[(size_t)b * NT_ + t0 + tid];
  async_wait_all();
  __syncthreads();

  // scores: S = (Q . K) / max(|q||k|, eps) + pos_bias   (64 tiles of 16x16)
  for (int tIdx = wave; tIdx < 64; tIdx += 8) {
    const int m0 = (tIdx & 1) * 16, c0 = (tIdx >> 1) * 16;
    v8f s = {};
#pragma unroll
    for (int kk = 0; kk < DP_; kk += 32) {
      v16bf a  = frag_a(Ql + m0 * DP_ + kk, DP_, lane);
      v16bf bb = frag_b_nk(Kl + c0 * DP_ + kk, DP_, lane);
      s = wmma_bf16(a, bb, s);
    }
    const int n = lane & 15, kh = (lane >> 4) * 8;
    const float knv = knl[c0 + n], pb = pbl[c0 + n];
#pragma unroll
    for (int r = 0; r < 8; ++r) {
      const int m = m0 + kh + r;
      Sl[m * NC_ + c0 + n] = s[r] / fmaxf(qnl[m] * knv, EPS_) + pb;
    }
  }
  __syncthreads();

  // softmax per row: 8 lanes per row, shfl reduction within 8-lane groups
  {
    const int row = tid >> 3, sub = tid & 7;
    float* srow = Sl + row * NC_;
    float mx = -1e30f;
    for (int c = sub; c < NC_; c += 8) mx = fmaxf(mx, srow[c]);
    mx = fmaxf(mx, __shfl_xor(mx, 1, 8));
    mx = fmaxf(mx, __shfl_xor(mx, 2, 8));
    mx = fmaxf(mx, __shfl_xor(mx, 4, 8));
    float sum = 0.0f;
    for (int c = sub; c < NC_; c += 8) {
      const float e = __expf(srow[c] - mx);
      srow[c] = e; sum += e;
    }
    sum += __shfl_xor(sum, 1, 8);
    sum += __shfl_xor(sum, 2, 8);
    sum += __shfl_xor(sum, 4, 8);
    const float inv = 1.0f / sum;
    bf16* prow = Pl + row * NC_;
    for (int c = sub; c < NC_; c += 8) prow[c] = (bf16)(srow[c] * inv);
  }
  __syncthreads();

  // O = P @ V  (8 tiles, one per wave, K = 512)
  {
    const int m0 = (wave >> 2) * 16, n0 = (wave & 3) * 16;
    v8f o = {};
#pragma unroll 4
    for (int kk = 0; kk < NC_; kk += 32) {
      v16bf a  = frag_a(Pl + m0 * NC_ + kk, NC_, lane);
      v16bf bb = frag_b_nk(Vtl + n0 * NC_ + kk, NC_, lane);
      o = wmma_bf16(a, bb, o);
    }
    const int n = n0 + (lane & 15), kh = (lane >> 4) * 8;
#pragma unroll
    for (int r = 0; r < 8; ++r) Ol[(m0 + kh + r) * DP_ + n] = (bf16)o[r];
  }
  __syncthreads();

  // out = O @ R_w^T + R_b  (32 tiles of 16x16, K = 64)
  for (int tIdx = wave; tIdx < 32; tIdx += 8) {
    const int m0 = (tIdx & 1) * 16, e0 = (tIdx >> 1) * 16;
    v8f s = {};
#pragma unroll
    for (int kk = 0; kk < DP_; kk += 32) {
      v16bf a  = frag_a(Ol + m0 * DP_ + kk, DP_, lane);
      v16bf bb = frag_b_nk(WRl + e0 * DP_ + kk, DP_, lane);
      s = wmma_bf16(a, bb, s);
    }
    const int e = e0 + (lane & 15), kh = (lane >> 4) * 8;
    const float rb = rbl[e];
    float* orow = out + ((size_t)b * NT_ + t0) * EMB_;
#pragma unroll
    for (int r = 0; r < 8; ++r)
      orow[(size_t)(m0 + kh + r) * EMB_ + e] = s[r] + rb;
  }
}

extern "C" void kernel_launch(void* const* d_in, const int* in_sizes, int n_in,
                              void* d_out, int out_size, void* d_ws, size_t ws_size,
                              hipStream_t stream) {
  const int*   titems   = (const int*)d_in[0];
  const int*   citems   = (const int*)d_in[1];
  const float* tvec     = (const float*)d_in[2];
  const float* cvec     = (const float*)d_in[3];
  const float* At_w     = (const float*)d_in[4];
  const float* At_b     = (const float*)d_in[5];
  const float* Ac_w     = (const float*)d_in[6];
  const float* Ac_b     = (const float*)d_in[7];
  const float* Bc_w     = (const float*)d_in[8];
  const float* Bc_b     = (const float*)d_in[9];
  const float* pos_bias = (const float*)d_in[10];
  const float* R_w      = (const float*)d_in[11];
  const float* R_b      = (const float*)d_in[12];
  float* out = (float*)d_out;

  char* ws = (char*)d_ws;
  const size_t szW  = (size_t)DP_ * EMB_ * sizeof(bf16);        // 32768
  const size_t szQ  = (size_t)B_ * NT_ * DP_ * sizeof(bf16);    // 8 MiB
  bf16*  WA = (bf16*)(ws);
  bf16*  WK = (bf16*)(ws + szW);
  bf16*  WV = (bf16*)(ws + 2 * szW);
  bf16*  WR = (bf16*)(ws + 3 * szW);
  bf16*  Qb = (bf16*)(ws + 4 * szW);
  bf16*  Kb = (bf16*)(ws + 4 * szW + szQ);
  bf16*  Vt = (bf16*)(ws + 4 * szW + 2 * szQ);
  float* qn = (float*)(ws + 4 * szW + 3 * szQ);
  float* kn = (float*)(ws + 4 * szW + 3 * szQ + (size_t)B_ * NT_ * sizeof(float));

  k_prep<<<256, 256, 0, stream>>>(At_w, Ac_w, Bc_w, R_w, WA, WK, WV, WR);
  k_proj_q<<<B_ * (NT_ / ROWS_), 256, 49536, stream>>>(titems, tvec, WA, At_b, Qb, qn);
  k_proj_kv<<<B_ * (NC_ / ROWS_), 256, 82560, stream>>>(citems, cvec, WK, WV, Ac_b,
                                                        Bc_b, Kb, Vt, kn);
  k_att<<<B_ * (NT_ / ROWS_), 256, 275584, stream>>>(Qb, Kb, Vt, qn, kn, pos_bias,
                                                     WR, R_b, out);
}
